// SpanScorer_51092930953469
// MI455X (gfx1250) — matrix operand order repaired and checked by reading, not compiled, not run on hardware
//
#include <hip/hip_runtime.h>
#include <hip/hip_bf16.h>
#include <math.h>

// ---------------- CDNA5 WMMA types ----------------
typedef __attribute__((ext_vector_type(16))) __bf16          v16bf;
typedef __attribute__((ext_vector_type(8)))  float           v8f;
typedef __attribute__((ext_vector_type(8)))  unsigned short  v8u16;

union FragU { v16bf v; v8u16 h[2]; };

__device__ __forceinline__ v8f wmma_bf16(const FragU& a, const FragU& b, v8f c) {
  return __builtin_amdgcn_wmma_f32_16x16x32_bf16(
      false, a.v, false, b.v, (short)0, c, false, false);
}

__device__ __forceinline__ unsigned short f2bf(float f) {
  union { float f; unsigned u; } c; c.f = f;
  unsigned u = c.u;
  u = u + 0x7FFFu + ((u >> 16) & 1u);   // round-to-nearest-even
  return (unsigned short)(u >> 16);
}
__device__ __forceinline__ float bf2f(unsigned short b) {
  union { unsigned u; float f; } c; c.u = ((unsigned)b) << 16;
  return c.f;
}

// ---------------- packing / utility kernels ----------------
__global__ void k_pack_bf16(const float* __restrict__ src,
                            unsigned short* __restrict__ dst, int n) {
  int i = blockIdx.x * blockDim.x + threadIdx.x;
  if (i < n) dst[i] = f2bf(src[i]);
}
__global__ void k_fill_u16(unsigned short* __restrict__ dst, int n) {
  int i = blockIdx.x * blockDim.x + threadIdx.x;
  if (i < n) dst[i] = 0;
}
__global__ void k_copy_f32(const float* __restrict__ src,
                           float* __restrict__ dst, int n) {
  int i = blockIdx.x * blockDim.x + threadIdx.x;
  if (i < n) dst[i] = src[i];
}
__global__ void k_fill_f32(float* __restrict__ dst, int n) {
  int i = blockIdx.x * blockDim.x + threadIdx.x;
  if (i < n) dst[i] = 0.0f;
}

// ---------------- generic bf16 WMMA GEMM, 2x2 register-blocked ----------
// C[M x N] (fp32) = A[M x K] (bf16 row-major) * Bt[N x K]^T (bf16 row-major)
// Block = 128 thr (4 waves) -> 64x64 block tile; each wave owns 32x32
// (2x2 WMMA tiles): 4 wmma per 4 fragment loads per K-step.
// epilogue: v = c + bias[n]; sigmoid applied if sigflags bit (n>>8) set.
__global__ void k_gemm_bf16(const unsigned short* __restrict__ A, int lda,
                            const unsigned short* __restrict__ Bt, int ldb,
                            float* __restrict__ C, int ldc, int K,
                            const float* __restrict__ bias, unsigned sigflags) {
  const int lane = threadIdx.x & 31;
  const int wave = threadIdx.x >> 5;                    // 0..3
  const int m0 = blockIdx.x * 64 + (wave >> 1) * 32;
  const int n0 = blockIdx.y * 64 + (wave & 1) * 32;
  const int g   = lane >> 4;                            // half-wave group
  const int l15 = lane & 15;

  // A frags: rows m0+l15 and m0+16+l15; K runs at k0+g*8 and k0+16+g*8
  const unsigned short* pa0 = A + (size_t)(m0 + l15) * lda + g * 8;
  const unsigned short* pa1 = pa0 + (size_t)16 * lda;
  // B frags: columns n0+l15 and n0+16+l15 (rows of Bt); K runs at k0+g*8 and k0+16+g*8
  const unsigned short* pb0 = Bt + (size_t)(n0 + l15) * ldb + g * 8;
  const unsigned short* pb1 = pb0 + (size_t)16 * ldb;

  v8f acc00 = {}, acc01 = {}, acc10 = {}, acc11 = {};
#pragma unroll 2
  for (int k0 = 0; k0 < K; k0 += 32) {
    FragU a0, a1, b0, b1;
    a0.h[0] = *(const v8u16*)(pa0); a0.h[1] = *(const v8u16*)(pa0 + 16);
    a1.h[0] = *(const v8u16*)(pa1); a1.h[1] = *(const v8u16*)(pa1 + 16);
    b0.h[0] = *(const v8u16*)(pb0); b0.h[1] = *(const v8u16*)(pb0 + 16);
    b1.h[0] = *(const v8u16*)(pb1); b1.h[1] = *(const v8u16*)(pb1 + 16);
    pa0 += 32; pa1 += 32; pb0 += 32; pb1 += 32;
    __builtin_prefetch(pa0, 0, 1);                      // -> global_prefetch_b8
    __builtin_prefetch(pb0, 0, 1);
    acc00 = wmma_bf16(a0, b0, acc00);
    acc01 = wmma_bf16(a0, b1, acc01);
    acc10 = wmma_bf16(a1, b0, acc10);
    acc11 = wmma_bf16(a1, b1, acc11);
  }

  // store: C/D layout: VGPR j -> row +j+8*g, col +l15
  const int mb = m0 + 8 * g;
  v8f accs[4] = { acc00, acc01, acc10, acc11 };
#pragma unroll
  for (int t = 0; t < 4; ++t) {
    const int mm = mb + (t >> 1) * 16;
    const int nn = n0 + (t & 1) * 16 + l15;
    const float bia = bias ? bias[nn] : 0.0f;
    const bool  sig = (sigflags >> (nn >> 8)) & 1u;
#pragma unroll
    for (int j = 0; j < 8; ++j) {
      float vv = accs[t][j] + bia;
      if (sig) vv = 1.0f / (1.0f + __expf(-vv));
      C[(size_t)(mm + j) * ldc + nn] = vv;
    }
  }
}

// ---------------- sequential RRNN scans --------------------
// FU rows r = l*8+b (l in [0,1035)), cols: [0,256) f_fw (sigmoided),
// [256,512) u_fw, [512,768) f_bw, [768,1024) u_bw.
__global__ void k_scan(const float* __restrict__ FU,
                       float* __restrict__ h_fw, float* __restrict__ fa_fw,
                       float* __restrict__ h_bw, float* __restrict__ fa_bw) {
  int tid = blockIdx.x * blockDim.x + threadIdx.x;
  if (tid >= 4096) return;
  const int dir = tid >> 11;
  const int ch  = tid & 2047;
  const int b   = ch >> 8;
  const int s   = ch & 255;
  if (dir == 0) {
    float c = 0.0f, fa = 0.0f;
    for (int l = 0; l < 1035; ++l) {
      size_t r = (size_t)(l * 8 + b);
      float f = FU[r * 1024 + s];
      float u = FU[r * 1024 + 256 + s];
      c = f * c + (1.0f - f) * u;
      fa += logf(f);
      h_fw[r * 256 + s]  = c;
      fa_fw[r * 256 + s] = fa;
    }
  } else {
    float c = 0.0f, fa = 0.0f;
    for (int l = 1034; l >= 0; --l) {
      size_t r = (size_t)(l * 8 + b);
      float f = FU[r * 1024 + 512 + s];
      float u = FU[r * 1024 + 768 + s];
      c = f * c + (1.0f - f) * u;
      fa += logf(f);
      h_bw[r * 256 + s]  = c;
      fa_bw[r * 256 + s] = fa;
    }
  }
}

// ---------------- span repr + gate + tanh -> bf16 ----------
// SpanT layout: [n][i], n = (b*1024+l)*9+v, i in [0,512).
__global__ void k_span(const float* __restrict__ h_fw, const float* __restrict__ fa_fw,
                       const float* __restrict__ h_bw, const float* __restrict__ fa_bw,
                       const float* __restrict__ GT,   // gate at cols [0,512)
                       unsigned short* __restrict__ SpanT) {
  size_t idx = (size_t)blockIdx.x * blockDim.x + threadIdx.x;
  if (idx >= (size_t)73728 * 512) return;
  const int i  = (int)(idx & 511);
  const int n  = (int)(idx >> 9);
  const int v  = n % 9;
  const int bl = n / 9;
  const int b  = bl >> 10, l = bl & 1023;
  float r;
  if (i < 256) {                                    // forward spans
    const int s = i;
    size_t re = (size_t)((l + 10) * 8 + b) * 256 + s;
    size_t rw = (size_t)((l + 1 + v) * 8 + b) * 256 + s;
    r = h_fw[re] - h_fw[rw] * __expf(fa_fw[re] - fa_fw[rw]);
  } else {                                          // backward spans
    const int s = i - 256;
    size_t re = (size_t)((l + 11) * 8 + b) * 256 + s;
    size_t rw = (size_t)((l + 2 + v) * 8 + b) * 256 + s;
    r = h_bw[rw] - h_bw[re] * __expf(fa_bw[rw] - fa_bw[re]);
  }
  const float g = GT[(size_t)(l * 8 + b) * 1024 + i];
  SpanT[idx] = f2bf(tanhf(r * g));
}

// ---------------- fused r2 GEMM + tanh + scores, 2x2 blocked -----------
// scores[n] = sum_h w2[h]*tanh( (Wspan @ span)[h,n] + tok[bl,h] + b1[h] )
// Block 128 thr / 4 waves; wave owns 32 columns, loops 16x (2 h-tiles each).
__global__ void k_r2_scores(const unsigned short* __restrict__ Wspan, // 512x512 bf16
                            const unsigned short* __restrict__ SpanT, // 73728x512 bf16
                            const float* __restrict__ GT,             // tok at cols [512,1024)
                            const float* __restrict__ b1,
                            const float* __restrict__ w2,
                            float* __restrict__ scores, int N) {
  __shared__ float ssc[128];
  ssc[threadIdx.x] = 0.0f;
  __syncthreads();

  const int lane = threadIdx.x & 31;
  const int wave = threadIdx.x >> 5;
  const int n0   = blockIdx.x * 128 + wave * 32;
  const int g    = lane >> 4;
  const int l15  = lane & 15;
  const int nr0  = n0 + l15;
  const int nr1  = nr0 + 16;

  const unsigned short* pbB0 = SpanT + (size_t)nr0 * 512 + g * 8;
  const unsigned short* pbB1 = pbB0 + (size_t)16 * 512;
  const int bl0 = nr0 / 9, bl1 = nr1 / 9;
  const float* tokrow0 = GT + (size_t)((bl0 & 1023) * 8 + (bl0 >> 10)) * 1024 + 512;
  const float* tokrow1 = GT + (size_t)((bl1 & 1023) * 8 + (bl1 >> 10)) * 1024 + 512;

  float partial0 = 0.0f, partial1 = 0.0f;
  for (int ht = 0; ht < 16; ++ht) {
    const int h0 = ht * 32;
    const unsigned short* pa0 = Wspan + (size_t)(h0 + l15) * 512 + g * 8;
    const unsigned short* pa1 = pa0 + (size_t)16 * 512;
    const unsigned short* pb0 = pbB0;
    const unsigned short* pb1 = pbB1;
    v8f acc00 = {}, acc01 = {}, acc10 = {}, acc11 = {};
#pragma unroll 2
    for (int k0 = 0; k0 < 512; k0 += 32) {
      FragU a0, a1, b0, b1;
      a0.h[0] = *(const v8u16*)(pa0); a0.h[1] = *(const v8u16*)(pa0 + 16);
      a1.h[0] = *(const v8u16*)(pa1); a1.h[1] = *(const v8u16*)(pa1 + 16);
      b0.h[0] = *(const v8u16*)(pb0); b0.h[1] = *(const v8u16*)(pb0 + 16);
      b1.h[0] = *(const v8u16*)(pb1); b1.h[1] = *(const v8u16*)(pb1 + 16);
      pa0 += 32; pa1 += 32; pb0 += 32; pb1 += 32;
      acc00 = wmma_bf16(a0, b0, acc00);
      acc01 = wmma_bf16(a0, b1, acc01);
      acc10 = wmma_bf16(a1, b0, acc10);
      acc11 = wmma_bf16(a1, b1, acc11);
    }
    const int ha = h0 + 8 * g;        // rows of tiles from a0
    const int hb = ha + 16;           // rows of tiles from a1
#pragma unroll
    for (int j = 0; j < 8; ++j) {
      const int h_a = ha + j, h_b = hb + j;
      partial0 += w2[h_a] * tanhf(acc00[j] + tokrow0[h_a] + b1[h_a]);
      partial1 += w2[h_a] * tanhf(acc01[j] + tokrow1[h_a] + b1[h_a]);
      partial0 += w2[h_b] * tanhf(acc10[j] + tokrow0[h_b] + b1[h_b]);
      partial1 += w2[h_b] * tanhf(acc11[j] + tokrow1[h_b] + b1[h_b]);
    }
  }
  atomicAdd(&ssc[wave * 32 + l15], partial0);       // -> ds_add_f32
  atomicAdd(&ssc[wave * 32 + 16 + l15], partial1);
  __syncthreads();
  {
    int n = blockIdx.x * 128 + threadIdx.x;
    if (n < N) scores[n] = ssc[threadIdx.x];
  }
}

// ---------------- softmax over V=9 -------------------------
__global__ void k_softmax(const float* __restrict__ scores, float* __restrict__ dist) {
  int t = blockIdx.x * blockDim.x + threadIdx.x;
  if (t >= 8192) return;
  const int b = t >> 10, l = t & 1023;
  const float* sc = scores + (size_t)(b * 1024 + l) * 9;
  float m = sc[0];
#pragma unroll
  for (int v = 1; v < 9; ++v) m = fmaxf(m, sc[v]);
  float e[9], sum = 0.0f;
#pragma unroll
  for (int v = 0; v < 9; ++v) { e[v] = __expf(sc[v] - m); sum += e[v]; }
  const float inv = 1.0f / sum;
  float* d = dist + (size_t)(l * 8 + b) * 9;        // output layout (l,b,v)
#pragma unroll
  for (int v = 0; v < 9; ++v) d[v] = e[v] * inv;
}

// ---------------- context gather ---------------------------
__global__ void k_context(const float* __restrict__ dist,
                          const unsigned short* __restrict__ SpanT,
                          float* __restrict__ ctx) {
  __shared__ float sd[9];
  const int bl = blockIdx.x;                        // b*1024 + l
  const int b = bl >> 10, l = bl & 1023;
  if (threadIdx.x < 9) sd[threadIdx.x] = dist[(size_t)(l * 8 + b) * 9 + threadIdx.x];
  __syncthreads();
  const int h = threadIdx.x;                        // 512 threads
  const unsigned short* sp = SpanT + (size_t)bl * 9 * 512 + h;
  float acc = 0.0f;
#pragma unroll
  for (int v = 0; v < 9; ++v) acc += sd[v] * bf2f(sp[(size_t)v * 512]);
  ctx[(size_t)(l * 8 + b) * 512 + h] = acc;
}

// ---------------- host orchestration -----------------------
static inline int cdiv(int a, int b) { return (a + b - 1) / b; }

extern "C" void kernel_launch(void* const* d_in, const int* in_sizes, int n_in,
                              void* d_out, int out_size, void* d_ws, size_t ws_size,
                              hipStream_t stream) {
  (void)in_sizes; (void)n_in; (void)out_size; (void)ws_size;
  const float* x      = (const float*)d_in[0];
  const float* init_x = (const float*)d_in[1];
  const float* w_gate = (const float*)d_in[2];
  const float* b_gate = (const float*)d_in[3];
  const float* w1_span= (const float*)d_in[4];
  const float* w1_tok = (const float*)d_in[5];
  const float* b1     = (const float*)d_in[6];
  const float* w2     = (const float*)d_in[7];
  const float* wf_fw  = (const float*)d_in[8];
  const float* bf_fw  = (const float*)d_in[9];
  const float* wu_fw  = (const float*)d_in[10];
  const float* wf_bw  = (const float*)d_in[11];
  const float* bf_bw  = (const float*)d_in[12];
  const float* wu_bw  = (const float*)d_in[13];

  // workspace carve (256B aligned)
  char* ws = (char*)d_ws;
  auto alloc = [&](size_t bytes) -> char* {
    char* p = ws; ws += (bytes + 255) & ~(size_t)255; return p;
  };
  unsigned short* Xa16   = (unsigned short*)alloc(8320ull * 512 * 2);   // x_aug bf16, 40 pad rows
  unsigned short* Wfu16  = (unsigned short*)alloc(1024ull * 512 * 2);   // [wf_fw;wu_fw;wf_bw;wu_bw]
  unsigned short* Wgt16  = (unsigned short*)alloc(1024ull * 512 * 2);   // [w_gate;w1_token]
  unsigned short* Wsp16  = (unsigned short*)alloc(512ull  * 512 * 2);   // w1_span
  float*          biasFU = (float*)alloc(1024 * 4);
  float*          biasGT = (float*)alloc(1024 * 4);
  float*          FU     = (float*)alloc(8320ull * 1024 * 4);
  float*          GT     = (float*)alloc(8192ull * 1024 * 4);
  float*          h_fw   = (float*)alloc(8280ull * 256 * 4);
  float*          fa_fw  = (float*)alloc(8280ull * 256 * 4);
  float*          h_bw   = (float*)alloc(8280ull * 256 * 4);
  float*          fa_bw  = (float*)alloc(8280ull * 256 * 4);
  unsigned short* SpanT  = (unsigned short*)alloc(73728ull * 512 * 2);
  float*          scores = (float*)alloc(73728ull * 4);

  float* dist = (float*)d_out;                      // (l,b,v) 73728
  float* ctx  = (float*)d_out + 73728;              // (l,b,h) 4194304

  const int T = 256;
  // ---- pack inputs to bf16 ----
  k_pack_bf16<<<cdiv(45056, T), T, 0, stream>>>(init_x, Xa16, 45056);
  k_pack_bf16<<<cdiv(4194304, T), T, 0, stream>>>(x, Xa16 + 45056, 4194304);
  k_fill_u16 <<<cdiv(20480, T), T, 0, stream>>>(Xa16 + 8280ull * 512, 20480);
  k_pack_bf16<<<cdiv(131072, T), T, 0, stream>>>(wf_fw, Wfu16 + 0,      131072);
  k_pack_bf16<<<cdiv(131072, T), T, 0, stream>>>(wu_fw, Wfu16 + 131072, 131072);
  k_pack_bf16<<<cdiv(131072, T), T, 0, stream>>>(wf_bw, Wfu16 + 262144, 131072);
  k_pack_bf16<<<cdiv(131072, T), T, 0, stream>>>(wu_bw, Wfu16 + 393216, 131072);
  k_pack_bf16<<<cdiv(262144, T), T, 0, stream>>>(w_gate, Wgt16 + 0,      262144);
  k_pack_bf16<<<cdiv(262144, T), T, 0, stream>>>(w1_tok, Wgt16 + 262144, 262144);
  k_pack_bf16<<<cdiv(262144, T), T, 0, stream>>>(w1_span, Wsp16, 262144);
  // packed per-column biases
  k_copy_f32<<<1, 256, 0, stream>>>(bf_fw, biasFU + 0,   256);
  k_fill_f32<<<1, 256, 0, stream>>>(biasFU + 256, 256);
  k_copy_f32<<<1, 256, 0, stream>>>(bf_bw, biasFU + 512, 256);
  k_fill_f32<<<1, 256, 0, stream>>>(biasFU + 768, 256);
  k_copy_f32<<<2, 256, 0, stream>>>(b_gate, biasGT, 512);
  k_fill_f32<<<2, 256, 0, stream>>>(biasGT + 512, 512);

  // ---- GEMM 1: f/u projections (sigmoid on f blocks: bits 0 and 2) ----
  k_gemm_bf16<<<dim3(130, 16), 128, 0, stream>>>(Xa16, 512, Wfu16, 512,
                                                 FU, 1024, 512, biasFU, 0x5u);
  // ---- GEMM 2: gate/token (sigmoid on gate blocks: bits 0,1) ----
  k_gemm_bf16<<<dim3(128, 16), 128, 0, stream>>>(Xa16 + 88ull * 512, 512, Wgt16, 512,
                                                 GT, 1024, 512, biasGT, 0x3u);
  // ---- RRNN scans ----
  k_scan<<<16, 256, 0, stream>>>(FU, h_fw, fa_fw, h_bw, fa_bw);
  // ---- span repr + gate + tanh -> bf16 (transposed for WMMA-B) ----
  k_span<<<147456, 256, 0, stream>>>(h_fw, fa_fw, h_bw, fa_bw, GT, SpanT);
  // ---- fused w1_span GEMM + tanh + w2 reduction -> scores ----
  k_r2_scores<<<576, 128, 0, stream>>>(Wsp16, SpanT, GT, b1, w2, scores, 73728);
  // ---- softmax + context ----
  k_softmax<<<32, 256, 0, stream>>>(scores, dist);
  k_context<<<8192, 512, 0, stream>>>(dist, SpanT, ctx);
}
